// Net_70334384439482
// MI455X (gfx1250) — matrix operand order repaired and checked by reading, not compiled, not run on hardware
//
#include <hip/hip_runtime.h>
#include <hip/hip_bf16.h>

// ---- problem constants (match reference) ----
#define NN   100000   // nodes
#define FIN  512      // input features
#define HD   256      // hidden
#define CD   64       // classes
#define KSTEPS 10
#define ALPHA 0.1f

typedef __attribute__((ext_vector_type(16))) __bf16 v16bf;
typedef __attribute__((ext_vector_type(8)))  __bf16 v8bf;
typedef __attribute__((ext_vector_type(8)))  float  v8f;

// ============================================================
// Weight pre-transpose to bf16, K-major per output column, so each
// WMMA B-fragment is a contiguous 32B load per lane.
// w0t[n*FIN + k] = bf16(w0[k*HD + n]);  w1t[c*HD + k] = bf16(w1[k*CD + c])
// ============================================================
__global__ void prep_w0t(const float* __restrict__ w0, __bf16* __restrict__ w0t) {
    int idx = blockIdx.x * blockDim.x + threadIdx.x;
    if (idx < HD * FIN) {
        int n = idx / FIN, k = idx % FIN;
        w0t[idx] = (__bf16)w0[k * HD + n];
    }
}
__global__ void prep_w1t(const float* __restrict__ w1, __bf16* __restrict__ w1t) {
    int idx = blockIdx.x * blockDim.x + threadIdx.x;
    if (idx < CD * HD) {
        int c = idx / HD, k = idx % HD;
        w1t[idx] = (__bf16)w1[k * CD + c];
    }
}

// ============================================================
// Degree / symmetric GCN norm: deg = 1 (self loop) + #incoming,
// then dinv = rsqrt(deg)
// ============================================================
__global__ void deg_init(float* __restrict__ deg) {
    int i = blockIdx.x * blockDim.x + threadIdx.x;
    if (i < NN) deg[i] = 1.0f;
}
__global__ void deg_edges(const long long* __restrict__ ei, float* __restrict__ deg, int E) {
    int e = blockIdx.x * blockDim.x + threadIdx.x;
    if (e < E) {
        int dst = (int)ei[(size_t)E + e];
        atomicAdd(&deg[dst], 1.0f);
    }
}
__global__ void deg_rsqrt(float* __restrict__ deg) {
    int i = blockIdx.x * blockDim.x + threadIdx.x;
    if (i < NN) {
        float d = deg[i];
        deg[i] = (d > 0.0f) ? rsqrtf(d) : 0.0f;
    }
}

// ============================================================
// Fused MLP: h = relu(x@w0+b0)@w1 + b1, one 16-row tile per block.
// 8 waves: stage A each wave does cols [w*32, w*32+32) of the hidden
// layer via 2 WMMA accumulators; bf16 tile parked in LDS; stage B
// waves 0..3 each produce 16 output columns. Writes h and z(=h).
// ============================================================
__global__ __launch_bounds__(256) void mlp_fused(
    const float* __restrict__ x, const __bf16* __restrict__ w0t,
    const float* __restrict__ b0, const __bf16* __restrict__ w1t,
    const float* __restrict__ b1, float* __restrict__ h, float* __restrict__ z)
{
    __shared__ __bf16 Tlds[16 * HD];           // 8 KB bf16 hidden tile
    const int wave  = threadIdx.x >> 5;
    const int lane  = threadIdx.x & 31;
    const int m     = lane & 15;               // M (stage A/B row) or N column within 16
    const int khalf = lane >> 4;               // which K half this lane holds
    const int rowBase = blockIdx.x * 16;

    // ---------------- Stage A: hidden = relu(x @ w0 + b0) ----------------
    const int n0 = wave * 32;
    v8f acc0 = {}, acc1 = {};
    const float* xr = x + (size_t)(rowBase + m) * FIN + khalf * 16;

    for (int k0 = 0; k0 < FIN; k0 += 32) {
        // A fragment (16x32 bf16): lane m holds K = khalf*16 + {0..15}
        v16bf a;
        const float* pa = xr + k0;
#pragma unroll
        for (int j = 0; j < 8; ++j) {
            a[j]     = (__bf16)pa[j];          // K = k0 + khalf*16 + j
            a[j + 8] = (__bf16)pa[j + 8];      // K = k0 + khalf*16 + 8 + j
        }
        // B fragments (32x16 bf16): lane holds column n, K = khalf*16 + {0..15}
        const int kb = k0 + khalf * 16;
        v16bf bA = *(const v16bf*)(w0t + (size_t)(n0 + m)      * FIN + kb);
        v16bf bB = *(const v16bf*)(w0t + (size_t)(n0 + 16 + m) * FIN + kb);

        acc0 = __builtin_amdgcn_wmma_f32_16x16x32_bf16(false, a, false, bA, (short)0, acc0, false, false);
        acc1 = __builtin_amdgcn_wmma_f32_16x16x32_bf16(false, a, false, bB, (short)0, acc1, false, false);
    }
    // bias + relu -> LDS bf16.  D layout: lane, reg r -> (M = r + 8*khalf, N = m)
    {
        const float bias0 = b0[n0 + m];
        const float bias1 = b0[n0 + 16 + m];
#pragma unroll
        for (int r = 0; r < 8; ++r) {
            int mr = r + 8 * khalf;
            float v0 = fmaxf(acc0[r] + bias0, 0.0f);
            float v1 = fmaxf(acc1[r] + bias1, 0.0f);
            Tlds[mr * HD + n0 + m]      = (__bf16)v0;
            Tlds[mr * HD + n0 + 16 + m] = (__bf16)v1;
        }
    }
    __syncthreads();

    // ---------------- Stage B: out = T @ w1 + b1 (waves 0..3) ----------------
    if (wave < 4) {
        const int c0 = wave * 16;
        v8f acc = {};
        for (int k0 = 0; k0 < HD; k0 += 32) {
            const __bf16* tp = &Tlds[m * HD + k0 + khalf * 16];
            v8bf lo = *(const v8bf*)tp;
            v8bf hi = *(const v8bf*)(tp + 8);
            v16bf a = __builtin_shufflevector(lo, hi,
                0, 1, 2, 3, 4, 5, 6, 7, 8, 9, 10, 11, 12, 13, 14, 15);
            v16bf b = *(const v16bf*)(w1t + (size_t)(c0 + m) * HD + k0 + khalf * 16);
            acc = __builtin_amdgcn_wmma_f32_16x16x32_bf16(false, a, false, b, (short)0, acc, false, false);
        }
        const float bias = b1[c0 + m];
#pragma unroll
        for (int r = 0; r < 8; ++r) {
            int row = rowBase + r + 8 * khalf;
            int col = c0 + m;
            float v = acc[r] + bias;
            h[(size_t)row * CD + col] = v;
            z[(size_t)row * CD + col] = v;   // z_0 = h
        }
    }
}

// ============================================================
// APPNP propagation step kernels
// ============================================================
// agg = dinv^2 * z    (self-loop contribution)
__global__ void init_agg(const float* __restrict__ dinv, const float* __restrict__ z,
                         float* __restrict__ agg) {
    int idx = blockIdx.x * blockDim.x + threadIdx.x;
    if (idx < NN * CD) {
        float d = dinv[idx >> 6];
        agg[idx] = d * d * z[idx];
    }
}
// agg[dst] += dinv[src]*dinv[dst] * z[src]  -- 16 threads/edge, float4 each
__global__ void edge_prop(const long long* __restrict__ ei, const float* __restrict__ dinv,
                          const float* __restrict__ z, float* __restrict__ agg, int E) {
    int idx = blockIdx.x * blockDim.x + threadIdx.x;
    if (idx < E * 16) {
        int e = idx >> 4, j = idx & 15;
        int src = (int)ei[e];
        int dst = (int)ei[(size_t)E + e];
        float norm = dinv[src] * dinv[dst];
        const float4 v = *(const float4*)(z + (size_t)src * CD + j * 4);
        float* ap = agg + (size_t)dst * CD + j * 4;
        atomicAdd(ap + 0, norm * v.x);
        atomicAdd(ap + 1, norm * v.y);
        atomicAdd(ap + 2, norm * v.z);
        atomicAdd(ap + 3, norm * v.w);
    }
}
// z = (1-alpha)*agg + alpha*h
__global__ void combine(const float* __restrict__ agg, const float* __restrict__ h,
                        float* __restrict__ z) {
    int idx = blockIdx.x * blockDim.x + threadIdx.x;
    if (idx < NN * CD) {
        z[idx] = (1.0f - ALPHA) * agg[idx] + ALPHA * h[idx];
    }
}

// ============================================================
// Row-wise log_softmax over C=64: one wave32 per row, 2 cols/lane
// ============================================================
__global__ __launch_bounds__(256) void logsoftmax(const float* __restrict__ z,
                                                  float* __restrict__ out) {
    int wave = threadIdx.x >> 5, lane = threadIdx.x & 31;
    int row  = blockIdx.x * 8 + wave;
    if (row >= NN) return;
    float v0 = z[(size_t)row * CD + lane];
    float v1 = z[(size_t)row * CD + lane + 32];
    float mx = fmaxf(v0, v1);
#pragma unroll
    for (int off = 16; off > 0; off >>= 1) mx = fmaxf(mx, __shfl_xor(mx, off, 32));
    float s = expf(v0 - mx) + expf(v1 - mx);
#pragma unroll
    for (int off = 16; off > 0; off >>= 1) s += __shfl_xor(s, off, 32);
    float ls = logf(s);
    out[(size_t)row * CD + lane]      = v0 - mx - ls;
    out[(size_t)row * CD + lane + 32] = v1 - mx - ls;
}

// ============================================================
static inline size_t algn(size_t x) { return (x + 255) & ~(size_t)255; }

extern "C" void kernel_launch(void* const* d_in, const int* in_sizes, int n_in,
                              void* d_out, int out_size, void* d_ws, size_t ws_size,
                              hipStream_t stream) {
    const float*     x  = (const float*)d_in[0];
    const float*     w0 = (const float*)d_in[1];
    const float*     b0 = (const float*)d_in[2];
    const float*     w1 = (const float*)d_in[3];
    const float*     b1 = (const float*)d_in[4];
    const long long* ei = (const long long*)d_in[5];
    const int E = in_sizes[5] / 2;

    // workspace carve-out (~78 MB)
    char* ws = (char*)d_ws;
    float*  h    = (float*)ws;  ws += algn((size_t)NN * CD * 4);
    float*  z    = (float*)ws;  ws += algn((size_t)NN * CD * 4);
    float*  agg  = (float*)ws;  ws += algn((size_t)NN * CD * 4);
    float*  dinv = (float*)ws;  ws += algn((size_t)NN * 4);
    __bf16* w0t  = (__bf16*)ws; ws += algn((size_t)HD * FIN * 2);
    __bf16* w1t  = (__bf16*)ws; ws += algn((size_t)CD * HD * 2);

    const int TB = 256;
    // weights -> bf16, K-major
    prep_w0t<<<(HD * FIN + TB - 1) / TB, TB, 0, stream>>>(w0, w0t);
    prep_w1t<<<(CD * HD + TB - 1) / TB, TB, 0, stream>>>(w1, w1t);

    // GCN normalization
    deg_init <<<(NN + TB - 1) / TB, TB, 0, stream>>>(dinv);
    deg_edges<<<(E + TB - 1) / TB, TB, 0, stream>>>(ei, dinv, E);
    deg_rsqrt<<<(NN + TB - 1) / TB, TB, 0, stream>>>(dinv);

    // fused MLP (WMMA): h = relu(x@w0+b0)@w1+b1 ; z = h
    mlp_fused<<<NN / 16, TB, 0, stream>>>(x, w0t, b0, w1t, b1, h, z);

    // APPNP: z <- 0.9 * Ahat z + 0.1 * h, K times
    for (int k = 0; k < KSTEPS; ++k) {
        init_agg <<<(NN * CD + TB - 1) / TB, TB, 0, stream>>>(dinv, z, agg);
        edge_prop<<<((size_t)E * 16 + TB - 1) / TB, TB, 0, stream>>>(ei, dinv, z, agg, E);
        combine  <<<(NN * CD + TB - 1) / TB, TB, 0, stream>>>(agg, h, z);
    }

    // log_softmax -> d_out
    logsoftmax<<<NN / 8, TB, 0, stream>>>(z, (float*)d_out);
}
